// SwinTransformerBlock_24438363914430
// MI455X (gfx1250) — compile-verified, matrix operand order checked by reading, not run on hardware
//
#include <hip/hip_runtime.h>
#include <hip/hip_bf16.h>
#include <stdint.h>

// ---------------------------------------------------------------------------
// Swin Transformer block for MI455X (gfx1250), wave32 + WMMA f16->f32,
// async global->LDS staging (ASYNCcnt) for the GEMM B panels.
// ---------------------------------------------------------------------------

typedef __attribute__((ext_vector_type(16))) _Float16 v16h;
typedef __attribute__((ext_vector_type(8)))  _Float16 h8;
typedef __attribute__((ext_vector_type(8)))  float    v8f;

// Problem constants (fixed by setup_inputs)
constexpr int BB     = 8;
constexpr int HW     = 128;      // H = W = 128
constexpr int C      = 192;
constexpr int HEADS  = 6;
constexpr int HD     = 32;
constexpr int WSZ    = 8;        // window size
constexpr int SHIFT  = 4;
constexpr int NTOK   = BB * HW * HW;              // 131072 tokens
constexpr int NWIN   = BB * (HW/WSZ) * (HW/WSZ);  // 2048 windows
constexpr int WTOK   = WSZ * WSZ;                 // 64 tokens / window
constexpr int QKVN   = 3 * C;                     // 576
constexpr int FFN    = 4 * C;                     // 768
constexpr int KC     = 192;                       // K-chunk staged in LDS

// ---------------------------------------------------------------------------
// 16x32 f16 A-fragment loader (row-major source, ld in halves).
// Lanes 0-15: row M=lane, elems K0..7,K16..23; lanes 16-31: row M=lane-16,
// elems K8..15,K24..31.  Same layout serves the transposed-B operand.
// ---------------------------------------------------------------------------
static __device__ __forceinline__ v16h load_frag16(const _Float16* base, long ld,
                                                   long row0, int k0, int lane) {
  long r  = row0 + (lane & 15);
  int  kb = k0 + ((lane >> 4) << 3);
  const _Float16* p = base + r * ld + kb;
  union { v16h v; h8 h[2]; } u;
  u.h[0] = *(const h8*)(p);
  u.h[1] = *(const h8*)(p + 16);
  return u.v;
}

static __device__ __forceinline__ v8f wmma_f16(v16h a, v16h b, v8f c) {
  return __builtin_amdgcn_wmma_f32_16x16x32_f16(false, a, false, b, (short)0, c,
                                                false, false);
}

// async copy of one 16-byte chunk global -> LDS (tracked by ASYNCcnt)
static __device__ __forceinline__ void async_copy16(const _Float16* g, _Float16* lds) {
  unsigned ldsa = (unsigned)(uintptr_t)lds;   // low 32 bits = LDS byte offset
  asm volatile("global_load_async_to_lds_b128 %0, %1, off"
               :: "v"(ldsa), "v"(g) : "memory");
}
static __device__ __forceinline__ void async_wait0() {
  asm volatile("s_wait_asynccnt 0x0" ::: "memory");
}

// ---------------------------------------------------------------------------
// Weight convert + transpose:  Wt[n*K + k] = (f16) W[k*N + n]
// ---------------------------------------------------------------------------
__global__ __launch_bounds__(256) void k_convT(const float* __restrict__ W,
                                               _Float16* __restrict__ Wt,
                                               int K, int N) {
  int i = blockIdx.x * 256 + threadIdx.x;
  if (i >= K * N) return;
  int k = i / N, n = i % N;
  Wt[(long)n * K + k] = (_Float16)W[(long)k * N + n];
}

// ---------------------------------------------------------------------------
// LayerNorm over C=192, one wave (32 lanes) per token, 6 channels per lane.
// windowed!=0: write in shifted-window-gathered row order (roll -4 + partition)
// ---------------------------------------------------------------------------
__global__ __launch_bounds__(256) void k_layernorm(const float* __restrict__ x,
                                                   const float* __restrict__ s,
                                                   const float* __restrict__ b,
                                                   _Float16* __restrict__ out,
                                                   int windowed) {
  int token = blockIdx.x * 8 + (threadIdx.x >> 5);
  int lane  = threadIdx.x & 31;
  const float* row = x + (long)token * C;
  float vals[6], sum = 0.f, sq = 0.f;
  #pragma unroll
  for (int i = 0; i < 6; ++i) {
    float v = row[lane + i * 32];
    vals[i] = v; sum += v; sq += v * v;
  }
  #pragma unroll
  for (int off = 16; off; off >>= 1) {
    sum += __shfl_xor(sum, off);
    sq  += __shfl_xor(sq,  off);
  }
  float mu   = sum * (1.f / C);
  float var  = sq * (1.f / C) - mu * mu;
  float rstd = rsqrtf(var + 1e-5f);

  long dst = token;
  if (windowed) {
    int bb  = token >> 14;
    int pos = token & 16383;
    int h0 = pos >> 7, w0 = pos & 127;
    int hs = (h0 + HW - SHIFT) & 127;
    int wp = (w0 + HW - SHIFT) & 127;
    dst = ((long)((bb * 16 + (hs >> 3)) * 16 + (wp >> 3)) << 6) |
          ((hs & 7) << 3) | (wp & 7);
  }
  _Float16* o = out + dst * C;
  #pragma unroll
  for (int i = 0; i < 6; ++i) {
    int c = lane + i * 32;
    o[c] = (_Float16)((vals[i] - mu) * rstd * s[c] + b[c]);
  }
}

// ---------------------------------------------------------------------------
// Blocked WMMA GEMM: out = epilogue(A[MxK] @ Bt[NxK]^T + bias)
// Block = 8 waves, covers 256 rows x 64 cols.  Each wave: 32x64 (8 tiles).
// B panel staged to LDS in 64x192 chunks via global_load_async_to_lds_b128.
//   mode 0: f16 store    mode 1: f16 + exact GELU    mode 2: f32 + residual
// ---------------------------------------------------------------------------
__global__ __launch_bounds__(256) void k_gemm16(const _Float16* __restrict__ A,
                                                const _Float16* __restrict__ Bt,
                                                const float* __restrict__ bias,
                                                _Float16* __restrict__ out16,
                                                float* __restrict__ out32,
                                                const float* __restrict__ res,
                                                int M, int N, int K, int mode) {
  __shared__ _Float16 Bs[64 * KC];   // 24 KB

  int tid  = threadIdx.x;
  int lane = tid & 31;
  int wave = tid >> 5;
  int NB   = N >> 6;                 // n-blocks of 64
  int nb   = blockIdx.x % NB;
  int mb   = blockIdx.x / NB;
  int n0   = nb << 6;
  long m0  = ((long)mb << 8) + ((long)wave << 5);   // 32 rows per wave

  v8f acc[2][4] = {};

  for (int kc = 0; kc < K; kc += KC) {
    __syncthreads();                 // previous chunk fully consumed
    // stage Bs = Bt[n0..n0+64)[kc..kc+KC)  -- 1536 16-byte chunks, 6/thread
    #pragma unroll
    for (int t = 0; t < 6; ++t) {
      int i   = tid + t * 256;
      int row = i / (KC / 8);
      int cof = (i % (KC / 8)) * 8;
      async_copy16(Bt + (long)(n0 + row) * K + kc + cof, &Bs[row * KC + cof]);
    }
    async_wait0();
    __syncthreads();

    for (int k0 = 0; k0 < KC; k0 += 32) {
      v16h a0 = load_frag16(A, K, m0,      kc + k0, lane);
      v16h a1 = load_frag16(A, K, m0 + 16, kc + k0, lane);
      #pragma unroll
      for (int j = 0; j < 4; ++j) {
        v16h b = load_frag16(Bs, KC, j * 16, k0, lane);  // ds_load_b128
        acc[0][j] = wmma_f16(a0, b, acc[0][j]);
        acc[1][j] = wmma_f16(a1, b, acc[1][j]);
      }
    }
  }

  int hi = lane >> 4;
  #pragma unroll
  for (int j = 0; j < 4; ++j) {
    int col = n0 + j * 16 + (lane & 15);
    float bv = bias[col];
    #pragma unroll
    for (int i = 0; i < 2; ++i) {
      #pragma unroll
      for (int r = 0; r < 8; ++r) {
        long row = m0 + i * 16 + hi * 8 + r;
        float v = acc[i][j][r] + bv;
        if (mode == 1) v = 0.5f * v * (1.f + erff(v * 0.70710678118654752f));
        if (mode == 2) out32[row * N + col] = res[row * N + col] + v;
        else           out16[row * N + col] = (_Float16)v;
      }
    }
  }
}

// ---------------------------------------------------------------------------
// Windowed attention: one block (4 waves) per (window, head).
// ---------------------------------------------------------------------------
__global__ __launch_bounds__(128) void k_attention(const _Float16* __restrict__ qkv,
                                                   const float* __restrict__ rpb,
                                                   _Float16* __restrict__ aout) {
  int wh   = blockIdx.x;
  int win  = wh / HEADS, head = wh % HEADS;
  const _Float16* base = qkv + (long)win * WTOK * QKVN;
  const _Float16* qB = base + head * HD;
  const _Float16* kB = base + C + head * HD;
  const _Float16* vB = base + 2 * C + head * HD;

  __shared__ float    S[WTOK * WTOK];      // 16 KB
  __shared__ _Float16 P[WTOK * WTOK];      //  8 KB
  __shared__ _Float16 vT[HD * WTOK];       //  4 KB  (vT[d][t])

  int tid = threadIdx.x, lane = tid & 31, wave = tid >> 5;

  // stage V transposed into LDS
  for (int i = tid; i < WTOK * HD; i += 128) {
    int t = i >> 5, d = i & 31;
    vT[d * WTOK + t] = vB[(long)t * QKVN + d];
  }

  // S = scale * q k^T   (each wave: one 16-row strip)
  int m0 = wave * 16;
  const float scale = 0.17677669529663687f;  // 32^-0.5
  for (int n0 = 0; n0 < WTOK; n0 += 16) {
    v16h a = load_frag16(qB, QKVN, m0, 0, lane);
    v16h b = load_frag16(kB, QKVN, n0, 0, lane);
    v8f acc = {};
    acc = wmma_f16(a, b, acc);
    int hi = lane >> 4, col = n0 + (lane & 15);
    #pragma unroll
    for (int r = 0; r < 8; ++r)
      S[(m0 + hi * 8 + r) * WTOK + col] = acc[r] * scale;
  }
  __syncthreads();

  // softmax with relative-position bias; one thread per query row
  if (tid < WTOK) {
    int q = tid, rq = q >> 3, cq = q & 7;
    float mx = -1e30f;
    for (int m = 0; m < WTOK; ++m) {
      int idx = (rq - (m >> 3) + 7) * 15 + (cq - (m & 7) + 7);
      mx = fmaxf(mx, S[q * WTOK + m] + rpb[idx * HEADS + head]);
    }
    float sum = 0.f;
    for (int m = 0; m < WTOK; ++m) {
      int idx = (rq - (m >> 3) + 7) * 15 + (cq - (m & 7) + 7);
      float e = __expf(S[q * WTOK + m] + rpb[idx * HEADS + head] - mx);
      S[q * WTOK + m] = e; sum += e;
    }
    float inv = 1.f / sum;
    for (int m = 0; m < WTOK; ++m)
      P[q * WTOK + m] = (_Float16)(S[q * WTOK + m] * inv);
  }
  __syncthreads();

  // O = P @ V   (16x32 per wave, K=64 -> two WMMA steps)
  _Float16* o = aout + (long)win * WTOK * C + head * HD;
  for (int n0 = 0; n0 < HD; n0 += 16) {
    v8f acc = {};
    for (int k0 = 0; k0 < WTOK; k0 += 32) {
      v16h a = load_frag16(P,  WTOK, m0, k0, lane);
      v16h b = load_frag16(vT, WTOK, n0, k0, lane);
      acc = wmma_f16(a, b, acc);
    }
    int hi = lane >> 4, col = n0 + (lane & 15);
    #pragma unroll
    for (int r = 0; r < 8; ++r) {
      int t = m0 + hi * 8 + r;
      o[(long)t * C + col] = (_Float16)acc[r];
    }
  }
}

// ---------------------------------------------------------------------------
// Reverse window partition + roll(+4) + residual:  x1 = x + scatter(proj)
// ---------------------------------------------------------------------------
__global__ __launch_bounds__(256) void k_unshift_residual(const float* __restrict__ x,
                                                          const _Float16* __restrict__ proj,
                                                          float* __restrict__ x1) {
  long i = (long)blockIdx.x * 256 + threadIdx.x;
  if (i >= (long)NTOK * C) return;
  long tok = i / C; int c = (int)(i % C);
  int bb  = (int)(tok >> 14);
  int pos = (int)(tok & 16383);
  int h0 = pos >> 7, w0 = pos & 127;
  int hs = (h0 + HW - SHIFT) & 127;
  int wp = (w0 + HW - SHIFT) & 127;
  long src = ((long)((bb * 16 + (hs >> 3)) * 16 + (wp >> 3)) << 6) |
             ((hs & 7) << 3) | (wp & 7);
  x1[i] = x[i] + (float)proj[src * C + c];
}

// ---------------------------------------------------------------------------
// host-side launcher
// ---------------------------------------------------------------------------
extern "C" void kernel_launch(void* const* d_in, const int* in_sizes, int n_in,
                              void* d_out, int out_size, void* d_ws, size_t ws_size,
                              hipStream_t stream) {
  (void)in_sizes; (void)n_in; (void)out_size; (void)ws_size;
  const float* x      = (const float*)d_in[0];
  const float* qkv_w  = (const float*)d_in[1];
  const float* qkv_b  = (const float*)d_in[2];
  const float* proj_w = (const float*)d_in[3];
  const float* proj_b = (const float*)d_in[4];
  const float* rpb    = (const float*)d_in[5];
  const float* n1s    = (const float*)d_in[6];
  const float* n1b    = (const float*)d_in[7];
  const float* n2s    = (const float*)d_in[8];
  const float* n2b    = (const float*)d_in[9];
  const float* w1     = (const float*)d_in[10];
  const float* b1     = (const float*)d_in[11];
  const float* w2     = (const float*)d_in[12];
  const float* b2     = (const float*)d_in[13];
  float* out = (float*)d_out;

  // workspace layout (bytes)
  char* ws = (char*)d_ws;
  _Float16* qkvwT = (_Float16*)(ws + 0);             // 576x192 f16
  _Float16* projwT= (_Float16*)(ws + 221184);        // 192x192 f16
  _Float16* w1T   = (_Float16*)(ws + 294912);        // 768x192 f16
  _Float16* w2T   = (_Float16*)(ws + 589824);        // 192x768 f16
  const size_t oW = 884736;
  _Float16* xn16  = (_Float16*)(ws + oW);                       // 50.3 MB
  _Float16* aout16= (_Float16*)(ws + oW + 50331648);            // 50.3 MB
  float*    x1    = (float*)   (ws + oW);                       // overlays xn16+aout16
  _Float16* h16   = (_Float16*)(ws + oW + 100663296);           // 50.3 MB
  _Float16* big16 = (_Float16*)(ws + oW + 150994944);           // 201.3 MB

  auto gblocks = [](int M, int N) { return (M / 256) * (N / 64); };

  // 1) weights -> transposed f16
  k_convT<<<(C * QKVN + 255) / 256, 256, 0, stream>>>(qkv_w,  qkvwT, C,   QKVN);
  k_convT<<<(C * C    + 255) / 256, 256, 0, stream>>>(proj_w, projwT, C,   C);
  k_convT<<<(C * FFN  + 255) / 256, 256, 0, stream>>>(w1,     w1T,   C,   FFN);
  k_convT<<<(FFN * C  + 255) / 256, 256, 0, stream>>>(w2,     w2T,   FFN, C);

  // 2) LN1 fused with shifted-window gather -> xn16 (windowed order)
  k_layernorm<<<NTOK / 8, 256, 0, stream>>>(x, n1s, n1b, xn16, 1);

  // 3) QKV GEMM: big16 = xn16 @ qkv_w + b   (M=131072, N=576, K=192)
  k_gemm16<<<gblocks(NTOK, QKVN), 256, 0, stream>>>(
      xn16, qkvwT, qkv_b, big16, nullptr, nullptr, NTOK, QKVN, C, 0);

  // 4) windowed attention -> aout16
  k_attention<<<NWIN * HEADS, 128, 0, stream>>>(big16, rpb, aout16);

  // 5) proj GEMM -> h16 (temporary windowed proj output), bias fused
  k_gemm16<<<gblocks(NTOK, C), 256, 0, stream>>>(
      aout16, projwT, proj_b, h16, nullptr, nullptr, NTOK, C, C, 0);

  // 6) reverse shift + residual -> x1 (f32)
  k_unshift_residual<<<(int)(((long)NTOK * C + 255) / 256), 256, 0, stream>>>(
      x, h16, x1);

  // 7) LN2 -> h16
  k_layernorm<<<NTOK / 8, 256, 0, stream>>>(x1, n2s, n2b, h16, 0);

  // 8) MLP1 with exact GELU -> big16   (M=131072, N=768, K=192)
  k_gemm16<<<gblocks(NTOK, FFN), 256, 0, stream>>>(
      h16, w1T, b1, big16, nullptr, nullptr, NTOK, FFN, C, 1);

  // 9) MLP2 + residual -> d_out (f32)  (M=131072, N=192, K=768)
  k_gemm16<<<gblocks(NTOK, C), 256, 0, stream>>>(
      big16, w2T, b2, nullptr, out, x1, NTOK, C, FFN, 2);
}
